// M3DFEL_49306224558684
// MI455X (gfx1250) — matrix operand order repaired and checked by reading, not compiled, not run on hardware
//
#include <hip/hip_runtime.h>
#include <math.h>

typedef __attribute__((ext_vector_type(16))) _Float16 v16h;
typedef __attribute__((ext_vector_type(8)))  float    v8f;

#define BM 64
#define BN 64
#define BK2 64   // K staged per barrier (2 x 32 for the wmma K)

// ---------------------------------------------------------------------------
// WMMA GEMM: C[M,N] = s0 * (A[M,K] x B[N,K]^T) + s1*add[M,N] + bias[N]
// A,add,C are f32; B is pre-converted f16. Staged to LDS; A converted f32->f16
// during staging. Requires M%64==0, N%64==0, K%64==0 (all call sites comply).
// Block = 128 threads = 4 waves; wave w computes rows [w*16, w*16+16) x 64 N.
// Per barrier pair each wave issues 8 v_wmma_f32_16x16x32_f16; all fragment
// ds_loads for a 32-K sub-step are issued before the 4-WMMA chain so the
// scheduler can overlap LDS latency with the matrix pipe.
// ---------------------------------------------------------------------------
__global__ __launch_bounds__(128)
void gemm_f16_wmma(const float* __restrict__ A, int lda,
                   const _Float16* __restrict__ B, int ldb,
                   float* __restrict__ C, int ldc,
                   const float* __restrict__ add, int ldadd,
                   float s0, float s1,
                   const float* __restrict__ bias,
                   int K)
{
  __shared__ _Float16 As[BM][BK2];   // 8 KB
  __shared__ _Float16 Bs[BN][BK2];   // 8 KB
  const int tid  = threadIdx.x;
  const int wave = tid >> 5;         // wave32
  const int lane = tid & 31;
  const int half = lane >> 4;
  const int ml   = lane & 15;
  const int bm = blockIdx.y * BM;
  const int bn = blockIdx.x * BN;

  v8f zero = {};
  v8f acc[4];
  acc[0] = zero; acc[1] = zero; acc[2] = zero; acc[3] = zero;

  for (int kb = 0; kb < K; kb += BK2) {
    // A tile: 64x64 f32 -> f16, float2 (2 elem) chunks: 2048 chunks / 128 thr
    for (int i = tid; i < (BM * BK2) / 2; i += 128) {
      int r = i >> 5;            // 32 chunks per row
      int c = (i & 31) * 2;
      float2 t = *(const float2*)&A[(size_t)(bm + r) * lda + kb + c];
      As[r][c]     = (_Float16)t.x;
      As[r][c + 1] = (_Float16)t.y;
    }
    // B tile: 64x64 f16, float2 (4 half) chunks: 1024 chunks / 128 thr
    for (int i = tid; i < (BN * BK2) / 4; i += 128) {
      int r = i >> 4;            // 16 chunks per row
      int c = (i & 15) * 4;
      *(float2*)&Bs[r][c] =
          *(const float2*)&B[(size_t)(bn + r) * ldb + kb + c];
    }
    __syncthreads();

    #pragma unroll
    for (int kk = 0; kk < 2; kk++) {
      const int ko = kk * 32;
      // ---- issue ALL fragment loads first ----
      // A fragment (16x32 f16): lane = m + 16*half; VGPR v holds K pairs:
      // K = half*16 + 2v
      union { v16h v; _Float16 e[16]; } af;
      {
        int m = wave * 16 + ml;
        #pragma unroll
        for (int vv = 0; vv < 8; vv++) {
          int kofs = ko + half * 16 + 2 * vv;
          af.e[2 * vv]     = As[m][kofs];
          af.e[2 * vv + 1] = As[m][kofs + 1];
        }
      }
      // B fragments (32x16, Bmat[k,n] = B[n,k]): lane n holds K 0..15
      // (half=0) or 16..31 (half=1); VGPR v holds rows 2v,2v+1.
      union { v16h v; _Float16 e[16]; } bfm[4];
      #pragma unroll
      for (int nt = 0; nt < 4; nt++) {
        int n = nt * 16 + ml;
        #pragma unroll
        for (int vv = 0; vv < 8; vv++) {
          int kofs = ko + half * 16 + 2 * vv;
          bfm[nt].e[2 * vv]     = Bs[n][kofs];
          bfm[nt].e[2 * vv + 1] = Bs[n][kofs + 1];
        }
      }
      // ---- then the back-to-back WMMA chain ----
      #pragma unroll
      for (int nt = 0; nt < 4; nt++) {
        acc[nt] = __builtin_amdgcn_wmma_f32_16x16x32_f16(
            false, af.v, false, bfm[nt].v, (short)0, acc[nt], false, false);
      }
    }
    __syncthreads();
  }

  // C/D layout: VGPR r -> M = r (lanes 0-15) / r+8 (lanes 16-31), N = lane&15
  const int row0 = bm + wave * 16 + half * 8;
  #pragma unroll
  for (int nt = 0; nt < 4; nt++) {
    #pragma unroll
    for (int r = 0; r < 8; r++) {
      size_t m = (size_t)(row0 + r);
      size_t n = (size_t)(bn + nt * 16 + ml);
      float v = s0 * acc[nt][r];
      if (add)  v += s1 * add[m * (size_t)ldadd + n];
      if (bias) v += bias[n];
      C[m * (size_t)ldc + n] = v;
    }
  }
}

// f32 -> f16 bulk conversion (weights, once per call)
__global__ __launch_bounds__(256)
void cvt_f16_kernel(const float* __restrict__ src, _Float16* __restrict__ dst,
                    size_t n)
{
  size_t i = (size_t)blockIdx.x * 256 + threadIdx.x;
  if (i < n) dst[i] = (_Float16)src[i];
}

// ---------------------------------------------------------------------------
// Helpers: block reductions for blockDim == 256
// ---------------------------------------------------------------------------
__device__ inline float blockSum256(float v, float* red) {
  int tid = threadIdx.x;
  red[tid] = v; __syncthreads();
  for (int s = 128; s > 0; s >>= 1) {
    if (tid < s) red[tid] += red[tid + s];
    __syncthreads();
  }
  float r = red[0]; __syncthreads();
  return r;
}
__device__ inline float blockMax256(float v, float* red) {
  int tid = threadIdx.x;
  red[tid] = v; __syncthreads();
  for (int s = 128; s > 0; s >>= 1) {
    if (tid < s) red[tid] = fmaxf(red[tid], red[tid + s]);
    __syncthreads();
  }
  float r = red[0]; __syncthreads();
  return r;
}
__device__ inline float sigf(float x) { return 1.0f / (1.0f + expf(-x)); }

// ---------------------------------------------------------------------------
// a[v,w] = (softmax_row(relu(nv1 @ nv2))[v,w] + (v==w)) * 0.5  (written f16)
// ---------------------------------------------------------------------------
__global__ __launch_bounds__(256)
void adj_kernel(const float* __restrict__ nv1, const float* __restrict__ nv2,
                _Float16* __restrict__ a)
{
  __shared__ float row[512];
  __shared__ float red[256];
  int v = blockIdx.x, tid = threadIdx.x;
  for (int w = tid; w < 512; w += 256) {
    float s = 0.f;
    #pragma unroll 8
    for (int c = 0; c < 32; c++) s += nv1[v * 32 + c] * nv2[c * 512 + w];
    row[w] = fmaxf(s, 0.f);
  }
  __syncthreads();
  float r0 = row[tid], r1 = row[tid + 256];
  float mx = blockMax256(fmaxf(r0, r1), red);
  float e0 = expf(r0 - mx), e1 = expf(r1 - mx);
  float sm = blockSum256(e0 + e1, red);
  a[(size_t)v * 512 + tid] =
      (_Float16)((e0 / sm + (tid == v ? 1.f : 0.f)) * 0.5f);
  a[(size_t)v * 512 + tid + 256] =
      (_Float16)((e1 / sm + (tid + 256 == v ? 1.f : 0.f)) * 0.5f);
}

// coef: A1[o]=sum_c mp_w[o,c]*sc_w[c]; A2[o]=sum_c mp_w[o,32+c]*sc_w[c];
//       C2[o]=sum_c (mp_w[o,c]+mp_w[o,32+c])*sc_b[c] + mp_b[o]
__global__ void coef_kernel(const float* __restrict__ mp_w,
                            const float* __restrict__ mp_b,
                            const float* __restrict__ sc_w,
                            const float* __restrict__ sc_b,
                            float* __restrict__ coef)
{
  int o = threadIdx.x;
  if (o < 32) {
    float a1 = 0.f, a2 = 0.f, c2 = mp_b[o];
    for (int c = 0; c < 32; c++) {
      float w1 = mp_w[o * 64 + c], w2 = mp_w[o * 64 + 32 + c];
      a1 += w1 * sc_w[c];
      a2 += w2 * sc_w[c];
      c2 += (w1 + w2) * sc_b[c];
    }
    coef[o] = a1; coef[32 + o] = a2; coef[64 + o] = c2;
  }
}

// oe[b,t,v] = sum_{o,l} gelu(A1[o]*x[b,l,v]+A2[o]*m[b,l,v]+C2[o])*ec_w[t,o,l] + ec_b[t]
__global__ __launch_bounds__(256)
void oe_kernel(const float* __restrict__ x, const float* __restrict__ m,
               const float* __restrict__ coef, const float* __restrict__ ec_w,
               const float* __restrict__ ec_b, float* __restrict__ oe)
{
  __shared__ float sc[96];
  __shared__ float sec[512];   // ec_w: [4,32,4]
  if (threadIdx.x < 96) sc[threadIdx.x] = coef[threadIdx.x];
  for (int i = threadIdx.x; i < 512; i += 256) sec[i] = ec_w[i];
  __syncthreads();
  int idx = blockIdx.x * 256 + threadIdx.x;      // b*512 + v
  int b = idx >> 9, v = idx & 511;
  float xv[4], mv[4];
  #pragma unroll
  for (int l = 0; l < 4; l++) {
    size_t o = ((size_t)b * 4 + l) * 512 + v;
    xv[l] = x[o]; mv[l] = m[o];
  }
  float acc[4] = {ec_b[0], ec_b[1], ec_b[2], ec_b[3]};
  for (int o = 0; o < 32; o++) {
    float a1 = sc[o], a2 = sc[32 + o], c2 = sc[64 + o];
    #pragma unroll
    for (int l = 0; l < 4; l++) {
      float t = a1 * xv[l] + a2 * mv[l] + c2;
      float g = 0.5f * t * (1.f + erff(t * 0.70710678118654752f));
      #pragma unroll
      for (int t4 = 0; t4 < 4; t4++) acc[t4] += g * sec[t4 * 128 + o * 4 + l];
    }
  }
  #pragma unroll
  for (int t4 = 0; t4 < 4; t4++)
    oe[((size_t)b * 4 + t4) * 512 + v] = acc[t4];
}

// xg = LN(x+g)*gg+gb; gw = softmax(xg, axis=-1); rows of 512
__global__ __launch_bounds__(256)
void ln_add_softmax512(const float* __restrict__ x, const float* __restrict__ g,
                       const float* __restrict__ gg, const float* __restrict__ gb,
                       float* __restrict__ xg, float* __restrict__ gw)
{
  __shared__ float red[256];
  size_t row = blockIdx.x;
  int tid = threadIdx.x;
  const float* xr = x + row * 512;
  const float* gr = g + row * 512;
  float v0 = xr[tid] + gr[tid];
  float v1 = xr[tid + 256] + gr[tid + 256];
  float mean = blockSum256(v0 + v1, red) * (1.f / 512.f);
  float d0 = v0 - mean, d1 = v1 - mean;
  float var = blockSum256(d0 * d0 + d1 * d1, red) * (1.f / 512.f);
  float inv = rsqrtf(var + 1e-5f);
  float o0 = d0 * inv * gg[tid] + gb[tid];
  float o1 = d1 * inv * gg[tid + 256] + gb[tid + 256];
  xg[row * 512 + tid] = o0;
  xg[row * 512 + tid + 256] = o1;
  float mx = blockMax256(fmaxf(o0, o1), red);
  float e0 = expf(o0 - mx), e1 = expf(o1 - mx);
  float sm = blockSum256(e0 + e1, red);
  gw[row * 512 + tid] = e0 / sm;
  gw[row * 512 + tid + 256] = e1 / sm;
}

// LN over rows of 1024; if gate != null: out = gate * sigmoid(LN(in)*gg+gb)
__global__ __launch_bounds__(256)
void ln1024(const float* __restrict__ in, const float* __restrict__ gg,
            const float* __restrict__ gb, const float* __restrict__ gate,
            float* __restrict__ out)
{
  __shared__ float red[256];
  size_t row = blockIdx.x;
  int tid = threadIdx.x;
  const float* ir = in + row * 1024;
  float v[4];
  float s = 0.f;
  #pragma unroll
  for (int i = 0; i < 4; i++) { v[i] = ir[tid + i * 256]; s += v[i]; }
  float mean = blockSum256(s, red) * (1.f / 1024.f);
  float sv = 0.f;
  #pragma unroll
  for (int i = 0; i < 4; i++) { float d = v[i] - mean; sv += d * d; }
  float var = blockSum256(sv, red) * (1.f / 1024.f);
  float inv = rsqrtf(var + 1e-5f);
  #pragma unroll
  for (int i = 0; i < 4; i++) {
    int e = tid + i * 256;
    float o = (v[i] - mean) * inv * gg[e] + gb[e];
    if (gate) out[row * 1024 + e] = gate[row * 1024 + e] * sigf(o);
    else      out[row * 1024 + e] = o;
  }
}

// LSTM cell: gate order i,f,o,c,s at strides of 512 in z (ld 2560)
__global__ __launch_bounds__(256)
void lstm_cell(const float* __restrict__ z, const float* __restrict__ gw, int tt,
               float* __restrict__ hbuf, float* __restrict__ cbuf, int layer,
               float* __restrict__ ycat, int dir, int last)
{
  int idx = blockIdx.x * 256 + threadIdx.x;     // b*512 + o
  int b = idx >> 9, o = idx & 511;
  const float* zr = z + (size_t)b * 2560;
  float i_t = sigf(zr[o]);
  float f_t = sigf(zr[512 + o]);
  float o_t = sigf(zr[1024 + o]);
  float ch  = tanhf(zr[1536 + o]);
  float dw  = gw[((size_t)b * 4 + tt) * 512 + o];
  float s_t = sigf(zr[2048 + o]) * dw;
  size_t st = (size_t)layer * 1024 * 512 + idx;
  float c_t = f_t * cbuf[st] + i_t * ch * s_t;
  float h_t = o_t * tanhf(c_t);
  cbuf[st] = c_t;
  hbuf[st] = h_t;
  if (last) ycat[((size_t)b * 4 + tt) * 1024 + dir * 512 + o] = h_t;
}

__global__ void zero_kernel(float* __restrict__ p, size_t n) {
  size_t i = (size_t)blockIdx.x * 256 + threadIdx.x;
  if (i < n) p[i] = 0.f;
}

// Tiny-sequence attention: 4 tokens, one block per (b, head); blockDim == d
__global__ void attn_small(const float* __restrict__ qkv, float* __restrict__ out,
                           int E, int heads, int d, float scale)
{
  extern __shared__ float sm[];
  float* sq = sm;             // 4*d
  float* sk = sq + 4 * d;     // 4*d
  float* sv = sk + 4 * d;     // 4*d
  float* ss = sv + 4 * d;     // 16 (scores -> probs)
  int bh = blockIdx.x;
  int b = bh / heads, h = bh % heads;
  int dt = threadIdx.x;
  size_t rs = (size_t)3 * E;
  size_t base = (size_t)b * 4 * rs + (size_t)h * d + dt;
  #pragma unroll
  for (int n = 0; n < 4; n++) {
    size_t o = base + (size_t)n * rs;
    sq[n * d + dt] = qkv[o];
    sk[n * d + dt] = qkv[o + E];
    sv[n * d + dt] = qkv[o + 2 * E];
  }
  __syncthreads();
  if (dt < 16) {
    int n = dt >> 2, m2 = dt & 3;
    float s = 0.f;
    for (int j = 0; j < d; j++) s += sq[n * d + j] * sk[m2 * d + j];
    ss[dt] = s * scale;
  }
  __syncthreads();
  if (dt < 4) {
    float mx = ss[dt * 4];
    for (int m2 = 1; m2 < 4; m2++) mx = fmaxf(mx, ss[dt * 4 + m2]);
    float e[4], sum = 0.f;
    for (int m2 = 0; m2 < 4; m2++) { e[m2] = expf(ss[dt * 4 + m2] - mx); sum += e[m2]; }
    for (int m2 = 0; m2 < 4; m2++) ss[dt * 4 + m2] = e[m2] / sum;
  }
  __syncthreads();
  #pragma unroll
  for (int n = 0; n < 4; n++) {
    float o = 0.f;
    #pragma unroll
    for (int m2 = 0; m2 < 4; m2++) o += ss[n * 4 + m2] * sv[m2 * d + dt];
    out[((size_t)b * 4 + n) * E + h * d + dt] = o;
  }
}

// Conv1d(4 -> 1, k=3, pad=1) over the feature axis (1024)
__global__ __launch_bounds__(256)
void pwconv_kernel(const float* __restrict__ y2, const float* __restrict__ pw_w,
                   const float* __restrict__ pw_b, float* __restrict__ xfer)
{
  int idx = blockIdx.x * 256 + threadIdx.x;   // b*1024 + e
  int b = idx >> 10, e = idx & 1023;
  float acc = pw_b[0];
  #pragma unroll
  for (int i = 0; i < 4; i++)
    #pragma unroll
    for (int k = 0; k < 3; k++) {
      int ee = e + k - 1;
      if (ee >= 0 && ee < 1024)
        acc += y2[((size_t)b * 4 + i) * 1024 + ee] * pw_w[i * 3 + k];
    }
  xfer[idx] = acc;
}

__global__ __launch_bounds__(256)
void fc_kernel(const float* __restrict__ xfer, const float* __restrict__ w,
               const float* __restrict__ bias, float* __restrict__ out)
{
  int idx = blockIdx.x * 256 + threadIdx.x;   // b*7 + j
  if (idx >= 1024 * 7) return;
  int b = idx / 7, j = idx % 7;
  float acc = bias[j];
  for (int e = 0; e < 1024; e++)
    acc += xfer[(size_t)b * 1024 + e] * w[(size_t)j * 1024 + e];
  out[idx] = acc;
}

// ---------------------------------------------------------------------------
extern "C" void kernel_launch(void* const* d_in, const int* in_sizes, int n_in,
                              void* d_out, int out_size, void* d_ws, size_t ws_size,
                              hipStream_t stream)
{
  const float* x        = (const float*)d_in[0];
  const float* nodevec1 = (const float*)d_in[1];
  const float* nodevec2 = (const float*)d_in[2];
  const float* sc_w     = (const float*)d_in[3];
  const float* sc_b     = (const float*)d_in[4];
  const float* mp_w     = (const float*)d_in[5];
  const float* mp_b     = (const float*)d_in[6];
  const float* ec_w     = (const float*)d_in[7];
  const float* ec_b     = (const float*)d_in[8];
  const float* gbl_w    = (const float*)d_in[9];
  const float* gbl_b    = (const float*)d_in[10];
  const float* gbln_g   = (const float*)d_in[11];
  const float* gbln_b   = (const float*)d_in[12];
  const float* lstm_W   = (const float*)d_in[13];
  const float* lstm_Wb  = (const float*)d_in[14];
  const float* lstm_U   = (const float*)d_in[15];
  const float* lstm_Ub  = (const float*)d_in[16];
  const float* mha_in_w = (const float*)d_in[17];
  const float* mha_in_b = (const float*)d_in[18];
  const float* mha_out_w= (const float*)d_in[19];
  const float* mha_out_b= (const float*)d_in[20];
  const float* ln1_g    = (const float*)d_in[21];
  const float* ln1_b    = (const float*)d_in[22];
  const float* qkv_w    = (const float*)d_in[23];
  const float* dmin_g   = (const float*)d_in[24];
  const float* dmin_b   = (const float*)d_in[25];
  const float* pw_w     = (const float*)d_in[26];
  const float* pw_b     = (const float*)d_in[27];
  const float* fc_w     = (const float*)d_in[28];
  const float* fc_b     = (const float*)d_in[29];

  float* outp = (float*)d_out;                // [B,7] then x_fer [B,1024]
  float* xfer = outp + 1024 * 7;

  // ---- workspace partition ----
  float* w = (float*)d_ws;
  size_t off = 0;
  auto alloc = [&](size_t n) { float* p = w + off; off += n; return p; };
  float* coef  = alloc(128);
  float* ymix  = alloc((size_t)4096 * 512);
  float* oe    = alloc((size_t)4096 * 512);
  float* xg    = alloc((size_t)4096 * 512);
  float* gw    = alloc((size_t)4096 * 512);
  float* gbuf  = alloc((size_t)4096 * 512);
  float* hbuf  = alloc((size_t)4 * 1024 * 512);
  float* cbuf  = alloc((size_t)4 * 1024 * 512);
  float* zbuf  = alloc((size_t)1024 * 2560);
  float* ycat  = alloc((size_t)4096 * 1024);
  float* qkvb  = alloc((size_t)4096 * 3072);
  float* attnb = alloc((size_t)4096 * 1024);
  float* orib  = alloc((size_t)4096 * 1024);
  float* y2b   = alloc((size_t)4096 * 1024);
  // f16 weight pool (allocated in float units: 1 float = 2 halves)
  auto alloch = [&](size_t nhalf) {
    _Float16* p = (_Float16*)(w + off); off += (nhalf + 1) / 2; return p;
  };
  const size_t NW_LSTM = (size_t)2 * 4 * 5 * 512 * 512;   // 10,485,760
  _Float16* a_adj  = alloch((size_t)512 * 512);
  _Float16* gblw_h = alloch((size_t)512 * 512);
  _Float16* lstmW_h= alloch(NW_LSTM);
  _Float16* lstmU_h= alloch(NW_LSTM);
  _Float16* mhain_h= alloch((size_t)3072 * 1024);
  _Float16* mhaout_h=alloch((size_t)1024 * 1024);
  _Float16* qkvw_h = alloch((size_t)3072 * 1024);
  (void)ws_size; (void)n_in; (void)in_sizes; (void)out_size;

  auto cvt = [&](const float* src, _Float16* dst, size_t n) {
    cvt_f16_kernel<<<(int)((n + 255) / 256), 256, 0, stream>>>(src, dst, n);
  };
  auto gemm = [&](const float* A, int lda, const _Float16* B, int ldb,
                  float* C, int ldc, const float* add, int ldadd,
                  float s0, float s1, const float* bias, int M, int N, int K) {
    dim3 grid(N / 64, M / 64);
    gemm_f16_wmma<<<grid, 128, 0, stream>>>(A, lda, B, ldb, C, ldc,
                                            add, ldadd, s0, s1, bias, K);
  };

  // ---- one-time f32 -> f16 weight conversion ----
  cvt(gbl_w,     gblw_h,  (size_t)512 * 512);
  cvt(lstm_W,    lstmW_h, NW_LSTM);
  cvt(lstm_U,    lstmU_h, NW_LSTM);
  cvt(mha_in_w,  mhain_h, (size_t)3072 * 1024);
  cvt(mha_out_w, mhaout_h,(size_t)1024 * 1024);
  cvt(qkv_w,     qkvw_h,  (size_t)3072 * 1024);

  // ---- Graph block ----
  adj_kernel<<<512, 256, 0, stream>>>(nodevec1, nodevec2, a_adj);
  coef_kernel<<<1, 32, 0, stream>>>(mp_w, mp_b, sc_w, sc_b, coef);
  // m[b,l,v] = 0.3 * sum_w x[b,l,w]*a[v,w] + 0.7 * x[b,l,v]
  gemm(x, 512, a_adj, 512, ymix, 512, x, 512, 0.3f, 0.7f, nullptr, 4096, 512, 512);
  oe_kernel<<<(1024 * 512) / 256, 256, 0, stream>>>(x, ymix, coef, ec_w, ec_b, oe);
  // g = oe @ gbl_w^T + gbl_b
  gemm(oe, 512, gblw_h, 512, gbuf, 512, nullptr, 0, 1.f, 0.f, gbl_b, 4096, 512, 512);
  ln_add_softmax512<<<4096, 256, 0, stream>>>(x, gbuf, gbln_g, gbln_b, xg, gw);

  // ---- Bi-LSTM ----
  for (int dir = 0; dir < 2; dir++) {
    size_t hc = (size_t)4 * 1024 * 512;
    zero_kernel<<<(int)((hc + 255) / 256), 256, 0, stream>>>(hbuf, hc);
    zero_kernel<<<(int)((hc + 255) / 256), 256, 0, stream>>>(cbuf, hc);
    for (int step = 0; step < 4; step++) {
      int tt = (dir == 0) ? step : (3 - step);
      for (int l = 0; l < 4; l++) {
        const float* xin; int lda;
        if (l == 0) { xin = xg + (size_t)tt * 512; lda = 2048; }
        else        { xin = hbuf + (size_t)(l - 1) * 1024 * 512; lda = 512; }
        const _Float16* W  = lstmW_h + (size_t)(dir * 4 + l) * 5 * 512 * 512;
        const _Float16* U  = lstmU_h + (size_t)(dir * 4 + l) * 5 * 512 * 512;
        const float* Wb = lstm_Wb + (size_t)(dir * 4 + l) * 5 * 512;
        const float* Ub = lstm_Ub + (size_t)(dir * 4 + l) * 5 * 512;
        const float* hl = hbuf + (size_t)l * 1024 * 512;
        // z = xin @ W^T + Wb
        gemm(xin, lda, W, 512, zbuf, 2560, nullptr, 0, 1.f, 0.f, Wb, 1024, 2560, 512);
        // z += h @ U^T + Ub
        gemm(hl, 512, U, 512, zbuf, 2560, zbuf, 2560, 1.f, 1.f, Ub, 1024, 2560, 512);
        lstm_cell<<<(1024 * 512) / 256, 256, 0, stream>>>(
            zbuf, gw, tt, hbuf, cbuf, l, ycat, dir, (l == 3) ? 1 : 0);
      }
    }
  }

  // ---- MHA (heads=4, d=256) ----
  gemm(ycat, 1024, mhain_h, 1024, qkvb, 3072, nullptr, 0, 1.f, 0.f, mha_in_b,
       4096, 3072, 1024);
  {
    int d = 256;
    size_t shm = (size_t)(12 * d + 16) * sizeof(float);
    attn_small<<<1024 * 4, d, shm, stream>>>(qkvb, attnb, 1024, 4, d, 1.f / 16.f);
  }
  gemm(attnb, 1024, mhaout_h, 1024, y2b, 1024, nullptr, 0, 1.f, 0.f, mha_out_b,
       4096, 1024, 1024);
  ln1024<<<4096, 256, 0, stream>>>(y2b, ln1_g, ln1_b, nullptr, orib);

  // ---- MIL attention (heads=8, d=128, no bias) ----
  gemm(orib, 1024, qkvw_h, 1024, qkvb, 3072, nullptr, 0, 1.f, 0.f, nullptr,
       4096, 3072, 1024);
  {
    int d = 128;
    size_t shm = (size_t)(12 * d + 16) * sizeof(float);
    attn_small<<<1024 * 8, d, shm, stream>>>(qkvb, attnb, 1024, 8, d,
                                             0.08838834764831845f); // 128^-0.5
  }
  // y2 = ori * sigmoid(LN(z)*dmin_g + dmin_b)
  ln1024<<<4096, 256, 0, stream>>>(attnb, dmin_g, dmin_b, orib, y2b);

  // ---- pwconv + fc ----
  pwconv_kernel<<<(1024 * 1024) / 256, 256, 0, stream>>>(y2b, pw_w, pw_b, xfer);
  fc_kernel<<<(1024 * 7 + 255) / 256, 256, 0, stream>>>(xfer, fc_w, fc_b, outp);
}
